// Seq2Seq_63479616635343
// MI455X (gfx1250) — matrix operand order repaired and checked by reading, not compiled, not run on hardware
//
#include <hip/hip_runtime.h>
#include <hip/hip_bf16.h>

#define Bn   1024
#define Sn   64
#define Tn   64
#define En   128
#define Hn   256
#define INV  30
#define OUTV 70

typedef __attribute__((ext_vector_type(16))) _Float16 v16h;
typedef __attribute__((ext_vector_type(8)))  float    v8f;

union V16 { v16h v; uint4 q[2]; };

__device__ __forceinline__ float sigmoidf_(float x) { return 1.0f / (1.0f + __expf(-x)); }

// ---------------- elementwise utility kernels ----------------
__global__ void k_cvt_f16(const float* __restrict__ s, _Float16* __restrict__ d, int n) {
    int i = blockIdx.x * blockDim.x + threadIdx.x;
    if (i < n) d[i] = (_Float16)s[i];
}
__global__ void k_bias2(const float* __restrict__ a, const float* __restrict__ b,
                        float* __restrict__ d, int n) {
    int i = blockIdx.x * blockDim.x + threadIdx.x;
    if (i < n) d[i] = a[i] + b[i];
}
__global__ void k_zero16(_Float16* __restrict__ d, int n) {
    int i = blockIdx.x * blockDim.x + threadIdx.x;
    if (i < n) d[i] = (_Float16)0.0f;
}
__global__ void k_zero32(float* __restrict__ d, int n) {
    int i = blockIdx.x * blockDim.x + threadIdx.x;
    if (i < n) d[i] = 0.0f;
}
__global__ void k_acc_h(float* __restrict__ acc, const _Float16* __restrict__ s, int n) {
    int i = blockIdx.x * blockDim.x + threadIdx.x;
    if (i < n) acc[i] += (float)s[i];
}
__global__ void k_acc_f(float* __restrict__ acc, const float* __restrict__ s, int n) {
    int i = blockIdx.x * blockDim.x + threadIdx.x;
    if (i < n) acc[i] += s[i];
}
__global__ void k_out_t0(float* __restrict__ out) {
    int i = blockIdx.x * blockDim.x + threadIdx.x;
    if (i < Bn * OUTV) {
        int b = i / OUTV, v = i % OUTV;
        out[(size_t)b * Tn * OUTV + v] = 0.0f;
    }
}
// gather token embeddings: out[(step,b,e)] (f16) = tab[tok[b*tok_ld + step]][e]
__global__ void k_embed(const int* __restrict__ tok, int tok_ld, int steps,
                        const float* __restrict__ tab, int vocab,
                        _Float16* __restrict__ out) {
    int i = blockIdx.x * blockDim.x + threadIdx.x;
    int total = steps * Bn * En;
    if (i >= total) return;
    int e = i % En;
    int b = (i / En) % Bn;
    int s = i / (En * Bn);
    int t = tok[b * tok_ld + s];
    t = t < 0 ? 0 : (t >= vocab ? vocab - 1 : t);
    out[i] = (_Float16)tab[(size_t)t * En + e];
}

// ---------------- WMMA building block ----------------
// acc[j][g] += A_j @ W_g^T over K, where A_j is a 16x32 f16 fragment from rows
// (arow0 / arow1) of `src` (row stride == K), and W_g is the gate-g row panel
// of W ((4H, K) row-major, f16). One B fragment feeds two WMMAs (M-blocking),
// so per K-step: 4 A-loads + 8 B-loads for 8 WMMAs.
// A frag (16x32 f16): lane m = L&15; half e -> K = 16*(e>>3) + 8*(L>=16) + (e&7)
// B frag (32x16 f16): lane n = L&15; half e -> K = 16*(e>>3) + 8*(L>=16) + (e&7)
__device__ __forceinline__ void mm_panel(const _Float16* __restrict__ src,
                                         const _Float16* __restrict__ W,
                                         int K, int arow0, int arow1, int ncol, int hi,
                                         v8f acc[2][4]) {
    for (int k0 = 0; k0 < K; k0 += 32) {
        V16 a0, a1;
        const _Float16* p0 = src + (size_t)arow0 * K + k0 + hi * 8;
        const _Float16* p1 = src + (size_t)arow1 * K + k0 + hi * 8;
        a0.q[0] = *(const uint4*)(p0);
        a0.q[1] = *(const uint4*)(p0 + 16);
        a1.q[0] = *(const uint4*)(p1);
        a1.q[1] = *(const uint4*)(p1 + 16);
#pragma unroll
        for (int g = 0; g < 4; ++g) {
            V16 b;
            const _Float16* w = W + (size_t)(g * Hn + ncol) * K + k0 + hi * 8;
            b.q[0] = *(const uint4*)(w);
            b.q[1] = *(const uint4*)(w + 16);
            acc[0][g] = __builtin_amdgcn_wmma_f32_16x16x32_f16(
                false, a0.v, false, b.v, (short)0, acc[0][g], false, false);
            acc[1][g] = __builtin_amdgcn_wmma_f32_16x16x32_f16(
                false, a1.v, false, b.v, (short)0, acc[1][g], false, false);
        }
    }
}

// ---------------- fused LSTM step (WMMA) ----------------
// gates = [x_t, h_prev] @ [Wih ; Whh]^T + bias. One wave = 32x16 (batch x hidden)
// tile: 2 M-subtiles x 4 gate accumulators. Gate math + c/h update done in the
// WMMA C/D register layout: VGPR r -> M = r + 8*(L>=16); N = L&15.
__global__ __launch_bounds__(128) void k_lstm_step(
    const _Float16* __restrict__ xt,    // (Bn, in_dim)
    const _Float16* __restrict__ Wih,   // (4H, in_dim) row-major, f16
    const _Float16* __restrict__ Whh,   // (4H, H)      row-major, f16
    const float*    __restrict__ bias,  // (4H) = bih + bhh
    const _Float16* __restrict__ hprev, // (Bn, H)
    const float*    __restrict__ cprev, // (Bn, H)
    _Float16*       __restrict__ hout,  // (Bn, H)
    float*          __restrict__ cout,  // (Bn, H)  (may alias cprev)
    _Float16*       __restrict__ yt,    // (Bn, ldy) pre-offset to (t, col0)
    int ldy, int in_dim)
{
    const int lane = threadIdx.x & 31;
    const int wv   = blockIdx.x * (blockDim.x >> 5) + (threadIdx.x >> 5);
    const int mt   = wv >> 4;            // Bn/32 = 32 tiles of 32 batch rows
    const int nt   = wv & 15;            // Hn/16 = 16 tiles
    const int lm   = lane & 15;
    const int hi   = lane >> 4;
    const int arow0 = mt * 32 + lm;      // batch rows for the two A fragments
    const int arow1 = arow0 + 16;
    const int ncol  = nt * 16 + lm;      // hidden column for B loads / stores

    v8f acc[2][4] = {};

    mm_panel(xt,    Wih, in_dim, arow0, arow1, ncol, hi, acc);  // x_t  @ Wih^T
    mm_panel(hprev, Whh, Hn,     arow0, arow1, ncol, hi, acc);  // h_t-1@ Whh^T

    // ---- gate nonlinearity + state update, in WMMA C/D layout ----
    const float bi = bias[0 * Hn + ncol];
    const float bf = bias[1 * Hn + ncol];
    const float bg = bias[2 * Hn + ncol];
    const float bo = bias[3 * Hn + ncol];
#pragma unroll
    for (int j = 0; j < 2; ++j) {
#pragma unroll
        for (int r = 0; r < 8; ++r) {
            const int brow  = mt * 32 + j * 16 + hi * 8 + r;
            const size_t ix = (size_t)brow * Hn + ncol;
            float gi = sigmoidf_(acc[j][0][r] + bi);
            float gf = sigmoidf_(acc[j][1][r] + bf);
            float gg = tanhf(acc[j][2][r] + bg);
            float go = sigmoidf_(acc[j][3][r] + bo);
            float c  = gf * cprev[ix] + gi * gg;
            float h  = go * tanhf(c);
            cout[ix] = c;
            hout[ix] = (_Float16)h;
            yt[(size_t)brow * ldy + ncol] = (_Float16)h;
        }
    }
}

// ---------------- final FC: logits = Y @ fcW^T + fcb (WMMA, N padded to 80) ----------------
__global__ __launch_bounds__(128) void k_fc(
    const _Float16* __restrict__ Y,     // (63*Bn, Hn) f16
    const _Float16* __restrict__ W,     // (OUTV, Hn) f16
    const float*    __restrict__ bias,  // (OUTV)
    float*          __restrict__ out)   // (Bn, Tn, OUTV)
{
    const int lane = threadIdx.x & 31;
    const int wv   = blockIdx.x * (blockDim.x >> 5) + (threadIdx.x >> 5);
    const int NT   = 5;                  // ceil(80/16)
    const int mt   = wv / NT;
    const int nt   = wv % NT;
    const int lm   = lane & 15;
    const int hi   = lane >> 4;
    const int arow = mt * 16 + lm;
    const int ncol = nt * 16 + lm;

    v8f acc = {};
    for (int k0 = 0; k0 < Hn; k0 += 32) {
        V16 a;
        const _Float16* ap = Y + (size_t)arow * Hn + k0 + hi * 8;
        a.q[0] = *(const uint4*)(ap);
        a.q[1] = *(const uint4*)(ap + 16);
        V16 b;
        b.q[0] = make_uint4(0u, 0u, 0u, 0u);
        b.q[1] = make_uint4(0u, 0u, 0u, 0u);
        if (ncol < OUTV) {
            const _Float16* wp = W + (size_t)ncol * Hn + k0 + hi * 8;
            b.q[0] = *(const uint4*)(wp);
            b.q[1] = *(const uint4*)(wp + 16);
        }
        acc = __builtin_amdgcn_wmma_f32_16x16x32_f16(false, a.v, false, b.v, (short)0, acc, false, false);
    }
    if (ncol < OUTV) {
        const float bb = bias[ncol];
#pragma unroll
        for (int r = 0; r < 8; ++r) {
            int mrow = mt * 16 + hi * 8 + r;   // 0 .. 63*Bn-1
            int t    = mrow >> 10;             // step index 0..62
            int bidx = mrow & 1023;            // batch index
            out[(size_t)bidx * Tn * OUTV + (size_t)(t + 1) * OUTV + ncol] = acc[r] + bb;
        }
    }
}

// ---------------- host orchestration ----------------
extern "C" void kernel_launch(void* const* d_in, const int* in_sizes, int n_in,
                              void* d_out, int out_size, void* d_ws, size_t ws_size,
                              hipStream_t stream) {
    (void)in_sizes; (void)n_in; (void)out_size; (void)ws_size;

    const int*   src     = (const int*)d_in[0];
    const int*   tgt     = (const int*)d_in[1];
    const float* enc_emb = (const float*)d_in[2];

    const float *wih_f[9], *whh_f[9], *bih_f[9], *bhh_f[9];
    int in_dims[9]; // scan index: encoder l*2+dir (0..5), decoder 6+l (6..8)
    for (int l = 0; l < 3; ++l)
        for (int d = 0; d < 2; ++d) {
            int base = 3 + (l * 2 + d) * 4;
            int si = l * 2 + d;
            wih_f[si] = (const float*)d_in[base + 0];
            whh_f[si] = (const float*)d_in[base + 1];
            bih_f[si] = (const float*)d_in[base + 2];
            bhh_f[si] = (const float*)d_in[base + 3];
            in_dims[si] = (l == 0) ? En : 2 * Hn;
        }
    const float* dec_emb = (const float*)d_in[27];
    for (int l = 0; l < 3; ++l) {
        int base = 28 + l * 4;
        int si = 6 + l;
        wih_f[si] = (const float*)d_in[base + 0];
        whh_f[si] = (const float*)d_in[base + 1];
        bih_f[si] = (const float*)d_in[base + 2];
        bhh_f[si] = (const float*)d_in[base + 3];
        in_dims[si] = (l == 0) ? En : Hn;
    }
    const float* fcW = (const float*)d_in[40];
    const float* fcb = (const float*)d_in[41];

    // ---- carve workspace ----
    char* wp = (char*)d_ws;
    auto carve = [&](size_t bytes) -> void* {
        void* r = (void*)wp;
        wp += (bytes + 255) & ~(size_t)255;
        return r;
    };
    _Float16* X0  = (_Float16*)carve((size_t)Sn * Bn * 512 * sizeof(_Float16));
    _Float16* X1  = (_Float16*)carve((size_t)Sn * Bn * 512 * sizeof(_Float16));
    _Float16* hA  = (_Float16*)carve((size_t)Bn * Hn * sizeof(_Float16));
    _Float16* hB  = (_Float16*)carve((size_t)Bn * Hn * sizeof(_Float16));
    float* cbuf   = (float*)carve((size_t)Bn * Hn * sizeof(float));
    float* hsum   = (float*)carve((size_t)Bn * Hn * sizeof(float));
    float* csum   = (float*)carve((size_t)Bn * Hn * sizeof(float));
    _Float16* wih16[9]; _Float16* whh16[9]; float* bias9[9];
    for (int si = 0; si < 9; ++si) {
        wih16[si] = (_Float16*)carve((size_t)4 * Hn * in_dims[si] * sizeof(_Float16));
        whh16[si] = (_Float16*)carve((size_t)4 * Hn * Hn * sizeof(_Float16));
        bias9[si] = (float*)carve((size_t)4 * Hn * sizeof(float));
    }
    _Float16* fcW16 = (_Float16*)carve((size_t)OUTV * Hn * sizeof(_Float16));

    auto gsz = [](int n) { return dim3((unsigned)((n + 255) / 256)); };

    // ---- weight conversion (f32 -> f16) + bias combine ----
    for (int si = 0; si < 9; ++si) {
        int n1 = 4 * Hn * in_dims[si];
        k_cvt_f16<<<gsz(n1), 256, 0, stream>>>(wih_f[si], wih16[si], n1);
        int n2 = 4 * Hn * Hn;
        k_cvt_f16<<<gsz(n2), 256, 0, stream>>>(whh_f[si], whh16[si], n2);
        k_bias2<<<gsz(4 * Hn), 256, 0, stream>>>(bih_f[si], bhh_f[si], bias9[si], 4 * Hn);
    }
    k_cvt_f16<<<gsz(OUTV * Hn), 256, 0, stream>>>(fcW, fcW16, OUTV * Hn);

    // ---- encoder embedding: (S, B, E) f16 into X0 ----
    k_embed<<<gsz(Sn * Bn * En), 256, 0, stream>>>(src, Sn, Sn, enc_emb, INV, X0);

    const int nBH = Bn * Hn;
    k_zero32<<<gsz(nBH), 256, 0, stream>>>(hsum, nBH);
    k_zero32<<<gsz(nBH), 256, 0, stream>>>(csum, nBH);

    // step grid: (Bn/32)*(Hn/16) = 512 waves -> 128 blocks of 4 waves
    const dim3 stepGrid(128), stepBlk(128);

    // ---- encoder: 3 bidirectional layers ----
    _Float16* Xin  = X0;
    _Float16* Xout = X1;
    for (int l = 0; l < 3; ++l) {
        for (int d = 0; d < 2; ++d) {
            int si = l * 2 + d;
            k_zero16<<<gsz(nBH), 256, 0, stream>>>(hA, nBH);
            k_zero32<<<gsz(nBH), 256, 0, stream>>>(cbuf, nBH);
            for (int i = 0; i < Sn; ++i) {
                int t = d ? (Sn - 1 - i) : i;
                const _Float16* xt = Xin + (size_t)t * Bn * in_dims[si];
                _Float16* yt = Xout + (size_t)t * Bn * 512 + (size_t)d * Hn;
                const _Float16* hp = (i & 1) ? hB : hA;
                _Float16*       hn = (i & 1) ? hA : hB;
                k_lstm_step<<<stepGrid, stepBlk, 0, stream>>>(
                    xt, wih16[si], whh16[si], bias9[si], hp, cbuf, hn, cbuf,
                    yt, 512, in_dims[si]);
            }
            // Sn=64 even -> final h landed in hA
            if (l < 2)
                k_acc_h<<<gsz(nBH), 256, 0, stream>>>(hsum, hA, nBH);
            else
                k_acc_f<<<gsz(nBH), 256, 0, stream>>>(csum, cbuf, nBH);
        }
        _Float16* tp = Xin; Xin = Xout; Xout = tp;
    }
    // after loop: Xin = encoder output buffer (unused further), Xout is free

    // ---- decoder embedding: (T-1, B, E) f16 into the free buffer ----
    k_embed<<<gsz(63 * Bn * En), 256, 0, stream>>>(tgt, Tn, 63, dec_emb, OUTV, Xout);
    _Float16* Din  = Xout;
    _Float16* Dout = Xin;

    // ---- decoder: 3 layers ----
    for (int l = 0; l < 3; ++l) {
        int si = 6 + l;
        if (l == 0) {
            k_cvt_f16<<<gsz(nBH), 256, 0, stream>>>(hsum, hA, nBH); // h0 = h_sum
            k_zero32<<<gsz(nBH), 256, 0, stream>>>(cbuf, nBH);
            k_acc_f<<<gsz(nBH), 256, 0, stream>>>(cbuf, csum, nBH); // c0 = c_sum
        } else {
            k_zero16<<<gsz(nBH), 256, 0, stream>>>(hA, nBH);
            k_zero32<<<gsz(nBH), 256, 0, stream>>>(cbuf, nBH);
        }
        for (int i = 0; i < 63; ++i) {
            const _Float16* xt = Din + (size_t)i * Bn * in_dims[si];
            _Float16* yt = Dout + (size_t)i * Bn * Hn;
            const _Float16* hp = (i & 1) ? hB : hA;
            _Float16*       hn = (i & 1) ? hA : hB;
            k_lstm_step<<<stepGrid, stepBlk, 0, stream>>>(
                xt, wih16[si], whh16[si], bias9[si], hp, cbuf, hn, cbuf,
                yt, Hn, in_dims[si]);
        }
        _Float16* tp = Din; Din = Dout; Dout = tp;
    }
    // decoder output sequence now in Din: (63*Bn, Hn)

    // ---- final projection + t=0 zero slice ----
    float* outp = (float*)d_out;
    k_out_t0<<<gsz(Bn * OUTV), 256, 0, stream>>>(outp);
    // waves = (63*Bn/16) * 5 = 20160 -> 5040 blocks of 4 waves
    k_fc<<<dim3(5040), dim3(128), 0, stream>>>(Din, fcW16, fcb, outp);
}
